// Mamba_44727789420631
// MI455X (gfx1250) — matrix operand order repaired and checked by reading, not compiled, not run on hardware
//
#include <hip/hip_runtime.h>
#include <hip/hip_bf16.h>
#include <math.h>

// ---------------- model dims ----------------
#define B_    4
#define L_    1024
#define DIN_  13
#define DINP_ 32      // DIN padded to 32 for K of WMMA
#define H_    768
#define ICH   1536
#define I2    3072
#define NST   16
#define KC    4
#define RR    48
#define RPAD  64      // R padded to 64
#define SSMW  80      // R + 2N
#define NL_   8
#define DA_   128
#define RP_   4
#define MROWS (B_*L_) // 4096

typedef __bf16 bf16_t;
typedef __attribute__((ext_vector_type(8)))  __bf16 v8bf;
typedef __attribute__((ext_vector_type(16))) __bf16 v16bf;
typedef __attribute__((ext_vector_type(8)))  float  v8f;

// act codes for gemm epilogue
#define ACT_NONE     0
#define ACT_RELU     1
#define ACT_SOFTPLUS 2
#define ACT_TANH     3

__device__ __forceinline__ bf16_t f2bf(float f) {
  unsigned int u = __float_as_uint(f);
  u += 0x7FFFu + ((u >> 16) & 1u);          // round-to-nearest-even
  unsigned short s = (unsigned short)(u >> 16);
  bf16_t b;
  __builtin_memcpy(&b, &s, sizeof(s));
  return b;
}
__device__ __forceinline__ float siluf(float x) { return x / (1.f + __expf(-x)); }

// ---------------- fp32 -> bf16 convert (with optional zero column padding) ----
__global__ void cvt_pad_kernel(const float* __restrict__ src, bf16_t* __restrict__ dst,
                               size_t total, int cols, int colsPad) {
  size_t idx = (size_t)blockIdx.x * blockDim.x + threadIdx.x;
  if (idx >= total) return;
  size_t r = idx / colsPad;
  int    c = (int)(idx % colsPad);
  float v = (c < cols) ? src[r * cols + c] : 0.f;
  dst[idx] = f2bf(v);
}

__device__ __forceinline__ float gemm_epilogue(float val, const float* residual,
                                               size_t off, int act) {
  if (residual) val += residual[off];
  if      (act == ACT_RELU)     val = fmaxf(val, 0.f);
  else if (act == ACT_SOFTPLUS) val = (val > 20.f) ? val : log1pf(__expf(val));
  else if (act == ACT_TANH)     val = tanhf(val);
  return val;
}

// ---------------- WMMA bf16 GEMM: C[M,N] = act(A[M,KP] * W[N,KP]^T + bias + res)
// Register-blocked (MT*16 x NT*16 tile per wave), K templated so all fragment
// addresses are base + compile-time immediate offsets (no per-row pointer
// arrays, minimal VGPR pressure). v_wmma_f32_16x16x32_bf16, full-EXEC waves.
template <int MT, int NT, int KP>
__global__ __launch_bounds__(256) void
wmma_gemm_kernel(const bf16_t* __restrict__ A, const bf16_t* __restrict__ W,
                 const float* __restrict__ bias, const float* __restrict__ residual,
                 float* __restrict__ C, int M, int N, int act) {
  const int lane    = threadIdx.x & 31;
  const int wave    = threadIdx.x >> 5;
  const int groupsN = N / (16 * NT);
  const int groupsM = M / (16 * MT);
  const int g       = blockIdx.x * 8 + wave;      // 8 waves / block
  if (g >= groupsM * groupsN) return;             // wave-uniform guard
  const int gm = g / groupsN;
  const int gn = g % groupsN;
  const int r  = lane & 15;
  const int hi = lane >> 4;                       // 0 or 1

  // Single base pointers; all row strides are compile-time immediates.
  // A fragment: lane half selects K subblocks {0-15} / {16-31}
  const bf16_t* aPtr = A + (size_t)(gm * MT * 16 + r) * KP + (hi << 4);
  // B fragment: col(n) = tile*16+r ; 16 contiguous K starting at hi*16
  const bf16_t* wPtr = W + (size_t)(gn * NT * 16 + r) * KP + (hi << 4);

  v8f acc[MT][NT];
#pragma unroll
  for (int i = 0; i < MT; ++i)
#pragma unroll
    for (int j = 0; j < NT; ++j) acc[i][j] = (v8f){};

  for (int k = 0; k < KP; k += 32) {
    v16bf av[MT];
#pragma unroll
    for (int i = 0; i < MT; ++i) {
      v8bf a0 = *(const v8bf*)(aPtr + k + i * 16 * KP);
      v8bf a1 = *(const v8bf*)(aPtr + k + i * 16 * KP + 8);
#pragma unroll
      for (int e = 0; e < 8; ++e) { av[i][e] = a0[e]; av[i][e + 8] = a1[e]; }
    }
    v16bf bv[NT];
#pragma unroll
    for (int j = 0; j < NT; ++j) bv[j] = *(const v16bf*)(wPtr + k + j * 16 * KP);
    if (k + 32 < KP) {                            // gfx1250 global_prefetch_b8
      __builtin_prefetch(aPtr + k + 32, 0, 1);
      __builtin_prefetch(wPtr + k + 32, 0, 1);
    }
#pragma unroll
    for (int i = 0; i < MT; ++i)
#pragma unroll
      for (int j = 0; j < NT; ++j)
        acc[i][j] = __builtin_amdgcn_wmma_f32_16x16x32_bf16(false, av[i], false, bv[j],
                                                            (short)0, acc[i][j], false, false);
  }

  // C/D layout: VGPR v -> row (v + 8*hi), col = lane&15
#pragma unroll
  for (int i = 0; i < MT; ++i) {
    const int mBase = (gm * MT + i) * 16 + (hi << 3);
#pragma unroll
    for (int j = 0; j < NT; ++j) {
      const int nIdx = (gn * NT + j) * 16 + r;
      const float bb = bias ? bias[nIdx] : 0.f;
#pragma unroll
      for (int v = 0; v < 8; ++v) {
        size_t off = (size_t)(mBase + v) * N + nIdx;
        C[off] = gemm_epilogue(acc[i][j][v] + bb, residual, off, act);
      }
    }
  }
}

// Dynamic-K fallback (16x16 tile per wave)
__global__ __launch_bounds__(256) void
wmma_gemm_dyn(const bf16_t* __restrict__ A, const bf16_t* __restrict__ W,
              const float* __restrict__ bias, const float* __restrict__ residual,
              float* __restrict__ C, int M, int N, int Kp, int act) {
  const int lane   = threadIdx.x & 31;
  const int wave   = threadIdx.x >> 5;
  const int tilesN = N >> 4;
  const int tile   = blockIdx.x * 8 + wave;
  if (tile >= (M >> 4) * tilesN) return;
  const int tm = tile / tilesN, tn = tile % tilesN;
  const int r = lane & 15, hi = lane >> 4;
  const bf16_t* aRow = A + (size_t)(tm * 16 + r) * Kp + (hi << 4);
  const bf16_t* wRow = W + (size_t)(tn * 16 + r) * Kp + (hi << 4);
  v8f acc = {};
  for (int k = 0; k < Kp; k += 32) {
    v8bf  a0 = *(const v8bf*)(aRow + k);
    v8bf  a1 = *(const v8bf*)(aRow + k + 8);
    v16bf bv = *(const v16bf*)(wRow + k);
    v16bf av;
#pragma unroll
    for (int e = 0; e < 8; ++e) { av[e] = a0[e]; av[e + 8] = a1[e]; }
    acc = __builtin_amdgcn_wmma_f32_16x16x32_bf16(false, av, false, bv, (short)0, acc, false, false);
  }
  const int nIdx  = tn * 16 + r;
  const int mBase = tm * 16 + (hi << 3);
  const float bb = bias ? bias[nIdx] : 0.f;
#pragma unroll
  for (int v = 0; v < 8; ++v) {
    size_t off = (size_t)(mBase + v) * N + nIdx;
    C[off] = gemm_epilogue(acc[v] + bb, residual, off, act);
  }
}

// ---------------- RMSNorm: per-row, writes bf16 (and optional fp32) ----------
__global__ void rmsnorm_kernel(const float* __restrict__ x, const float* __restrict__ w,
                               bf16_t* __restrict__ obf, float* __restrict__ of32, int Hd) {
  __shared__ float red[256];
  const int m = blockIdx.x;
  const int tid = threadIdx.x;
  const float* row = x + (size_t)m * Hd;
  float s = 0.f;
  for (int j = tid; j < Hd; j += 256) { float v = row[j]; s += v * v; }
  red[tid] = s; __syncthreads();
  for (int st = 128; st > 0; st >>= 1) { if (tid < st) red[tid] += red[tid + st]; __syncthreads(); }
  float rs = rsqrtf(red[0] / (float)Hd + 1e-5f);
  for (int j = tid; j < Hd; j += 256) {
    float v = row[j] * rs * w[j];
    obf[(size_t)m * Hd + j] = f2bf(v);
    if (of32) of32[(size_t)m * Hd + j] = v;
  }
}

// ---------------- causal depthwise conv (K=4) + SiLU ------------------------
__global__ void conv_silu_kernel(const float* __restrict__ proj, const float* __restrict__ cw,
                                 const float* __restrict__ cb, float* __restrict__ u,
                                 bf16_t* __restrict__ ubf) {
  int idx = blockIdx.x * blockDim.x + threadIdx.x;     // over MROWS*ICH
  if (idx >= MROWS * ICH) return;
  int c = idx % ICH;
  int m = idx / ICH;                                   // b*L + l
  int l = m % L_;
  float acc = cb[c];
#pragma unroll
  for (int k = 0; k < KC; ++k) {
    int lp = l - (KC - 1) + k;
    if (lp >= 0) acc += cw[c * KC + k] * proj[(size_t)(m - (KC - 1) + k) * I2 + c];
  }
  float v = siluf(acc);
  u[idx] = v;
  ubf[idx] = f2bf(v);
}

// ---------------- build padded bf16 ts from ssm[:, :R] ----------------------
__global__ void ts_pad_kernel(const float* __restrict__ ssm, bf16_t* __restrict__ ts) {
  int idx = blockIdx.x * blockDim.x + threadIdx.x;     // MROWS*RPAD
  if (idx >= MROWS * RPAD) return;
  int m = idx / RPAD, c = idx % RPAD;
  float v = (c < RR) ? ssm[(size_t)m * SSMW + c] : 0.f;
  ts[idx] = f2bf(v);
}

// ---------------- selective scan: thread per (b, channel) -------------------
__global__ void scan_kernel(const float* __restrict__ dt, const float* __restrict__ u,
                            const float* __restrict__ ssm, const float* __restrict__ alog,
                            const float* __restrict__ Dp, float* __restrict__ y) {
  int idx = blockIdx.x * blockDim.x + threadIdx.x;     // B_*ICH
  if (idx >= B_ * ICH) return;
  int i = idx % ICH;
  int b = idx / ICH;
  float A[NST], h[NST];
#pragma unroll
  for (int n = 0; n < NST; ++n) { A[n] = -__expf(alog[(size_t)i * NST + n]); h[n] = 0.f; }
  const float Dv = Dp[i];
  for (int l = 0; l < L_; ++l) {
    size_t m = (size_t)b * L_ + l;
    float dtv = dt[m * ICH + i];
    float uv  = u[m * ICH + i];
    float du  = dtv * uv;
    const float* Bt = ssm + m * SSMW + RR;
    const float* Ct = Bt + NST;
    float acc = 0.f;
#pragma unroll
    for (int n = 0; n < NST; ++n) {
      h[n] = __expf(dtv * A[n]) * h[n] + du * Bt[n];
      acc += h[n] * Ct[n];
    }
    y[m * ICH + i] = acc + uv * Dv;
  }
}

// ---------------- y * silu(gate) -> bf16 ------------------------------------
__global__ void gate_mul_kernel(const float* __restrict__ y, const float* __restrict__ proj,
                                bf16_t* __restrict__ ybf) {
  int idx = blockIdx.x * blockDim.x + threadIdx.x;     // MROWS*ICH
  if (idx >= MROWS * ICH) return;
  int i = idx % ICH;
  int m = idx / ICH;
  float g = proj[(size_t)m * I2 + ICH + i];
  ybf[idx] = f2bf(y[idx] * siluf(g));
}

// ---------------- softmax over L per (b, r) ---------------------------------
__global__ void softmax_l_kernel(const float* __restrict__ logits, float* __restrict__ attnw) {
  __shared__ float red[256];
  const int b = blockIdx.x >> 2;
  const int r = blockIdx.x & 3;
  const int tid = threadIdx.x;
  float mx = -3.4e38f;
  for (int l = tid; l < L_; l += 256) mx = fmaxf(mx, logits[((size_t)b * L_ + l) * RP_ + r]);
  red[tid] = mx; __syncthreads();
  for (int s = 128; s > 0; s >>= 1) { if (tid < s) red[tid] = fmaxf(red[tid], red[tid + s]); __syncthreads(); }
  mx = red[0]; __syncthreads();
  float sum = 0.f;
  for (int l = tid; l < L_; l += 256) sum += __expf(logits[((size_t)b * L_ + l) * RP_ + r] - mx);
  red[tid] = sum; __syncthreads();
  for (int s = 128; s > 0; s >>= 1) { if (tid < s) red[tid] += red[tid + s]; __syncthreads(); }
  float inv = 1.f / red[0];
  for (int l = tid; l < L_; l += 256) {
    size_t o = ((size_t)b * L_ + l) * RP_ + r;
    attnw[o] = __expf(logits[o] - mx) * inv;
  }
}

// ---------------- attention pooling: m[b,r,h] = sum_l attn*hf ---------------
__global__ void pool_kernel(const float* __restrict__ attnw, const float* __restrict__ hf,
                            float* __restrict__ mpool) {
  int idx = blockIdx.x * blockDim.x + threadIdx.x;     // B_*RP_*H_
  if (idx >= B_ * RP_ * H_) return;
  int hh = idx % H_;
  int r  = (idx / H_) % RP_;
  int b  = idx / (H_ * RP_);
  float s = 0.f;
  for (int l = 0; l < L_; ++l) {
    size_t m = (size_t)b * L_ + l;
    s += attnw[m * RP_ + r] * hf[m * H_ + hh];
  }
  mpool[idx] = s;
}

// ---------------- tiny scalar GEMM (M small): C = act(A*W^T + bias) ---------
__global__ void small_gemm_kernel(const float* __restrict__ A, const float* __restrict__ W,
                                  const float* __restrict__ bias, float* __restrict__ C,
                                  int M, int N, int K, int act) {
  int idx = blockIdx.x * blockDim.x + threadIdx.x;
  if (idx >= M * N) return;
  int m = idx / N, n = idx % N;
  float s = bias ? bias[n] : 0.f;
  for (int k = 0; k < K; ++k) s += A[(size_t)m * K + k] * W[(size_t)n * K + k];
  if (act == ACT_RELU) s = fmaxf(s, 0.f);
  C[idx] = s;
}

// =============================================================================
extern "C" void kernel_launch(void* const* d_in, const int* in_sizes, int n_in,
                              void* d_out, int out_size, void* d_ws, size_t ws_size,
                              hipStream_t stream) {
  (void)in_sizes; (void)n_in; (void)out_size; (void)ws_size;
  const float* x      = (const float*)d_in[0];
  const float* w1     = (const float*)d_in[1];
  const float* b1     = (const float*)d_in[2];
  const float* w2     = (const float*)d_in[3];
  const float* b2     = (const float*)d_in[4];
  const float* w3     = (const float*)d_in[5];
  const float* b3     = (const float*)d_in[6];
  const float* norm_w = (const float*)d_in[7];
  const float* inw    = (const float*)d_in[8];
  const float* convw  = (const float*)d_in[9];
  const float* convb  = (const float*)d_in[10];
  const float* xprojw = (const float*)d_in[11];
  const float* dtw    = (const float*)d_in[12];
  const float* dtb    = (const float*)d_in[13];
  const float* alog   = (const float*)d_in[14];
  const float* Dp     = (const float*)d_in[15];
  const float* outw   = (const float*)d_in[16];
  const float* normf  = (const float*)d_in[17];
  const float* aw1    = (const float*)d_in[18];
  const float* aw2    = (const float*)d_in[19];
  const float* ow1    = (const float*)d_in[20];
  const float* ob1    = (const float*)d_in[21];
  const float* ow2    = (const float*)d_in[22];
  const float* ob2    = (const float*)d_in[23];
  const float* ow3    = (const float*)d_in[24];
  const float* ob3    = (const float*)d_in[25];
  float* out = (float*)d_out;

  // ---- bump allocator over workspace ----
  char* p = (char*)d_ws;
  auto alloc = [&](size_t bytes) -> void* {
    void* r = (void*)p; p += (bytes + 255) & ~(size_t)255; return r;
  };

  // bf16 weight/activation buffers
  bf16_t* x_bf      = (bf16_t*)alloc((size_t)MROWS * DINP_ * 2);
  bf16_t* w1_bf     = (bf16_t*)alloc((size_t)256 * DINP_ * 2);
  bf16_t* w2_bf     = (bf16_t*)alloc((size_t)256 * 256 * 2);
  bf16_t* w3_bf     = (bf16_t*)alloc((size_t)H_ * 256 * 2);
  bf16_t* inw_bf    = (bf16_t*)alloc((size_t)NL_ * I2 * H_ * 2);
  bf16_t* xprojw_bf = (bf16_t*)alloc((size_t)NL_ * SSMW * ICH * 2);
  bf16_t* dtw_bf    = (bf16_t*)alloc((size_t)NL_ * ICH * RPAD * 2);
  bf16_t* outw_bf   = (bf16_t*)alloc((size_t)NL_ * H_ * ICH * 2);
  bf16_t* aw1_bf    = (bf16_t*)alloc((size_t)DA_ * H_ * 2);
  // fp32/bf16 activations
  float*  t256a     = (float*)alloc((size_t)MROWS * 256 * 4);
  bf16_t* t256a_bf  = (bf16_t*)alloc((size_t)MROWS * 256 * 2);
  float*  t256b     = (float*)alloc((size_t)MROWS * 256 * 4);
  bf16_t* t256b_bf  = (bf16_t*)alloc((size_t)MROWS * 256 * 2);
  float*  h         = (float*)alloc((size_t)MROWS * H_ * 4);
  bf16_t* hn_bf     = (bf16_t*)alloc((size_t)MROWS * H_ * 2);
  float*  proj      = (float*)alloc((size_t)MROWS * I2 * 4);
  float*  u         = (float*)alloc((size_t)MROWS * ICH * 4);
  bf16_t* u_bf      = (bf16_t*)alloc((size_t)MROWS * ICH * 2);
  float*  ssm       = (float*)alloc((size_t)MROWS * SSMW * 4);
  bf16_t* ts_bf     = (bf16_t*)alloc((size_t)MROWS * RPAD * 2);
  float*  dt        = (float*)alloc((size_t)MROWS * ICH * 4);
  float*  y         = (float*)alloc((size_t)MROWS * ICH * 4);
  bf16_t* y_bf      = (bf16_t*)alloc((size_t)MROWS * ICH * 2);
  float*  hf        = (float*)alloc((size_t)MROWS * H_ * 4);
  bf16_t* hf_bf     = (bf16_t*)alloc((size_t)MROWS * H_ * 2);
  float*  s1        = (float*)alloc((size_t)MROWS * DA_ * 4);
  float*  logits    = (float*)alloc((size_t)MROWS * RP_ * 4);
  float*  attnw     = (float*)alloc((size_t)MROWS * RP_ * 4);
  float*  mpool     = (float*)alloc((size_t)B_ * RP_ * H_ * 4);
  float*  o1        = (float*)alloc((size_t)B_ * 256 * 4);
  float*  o2        = (float*)alloc((size_t)B_ * 256 * 4);

  auto cvt = [&](const float* src, bf16_t* dst, size_t rows, int cols, int colsPad) {
    size_t total = rows * (size_t)colsPad;
    cvt_pad_kernel<<<dim3((unsigned)((total + 255) / 256)), dim3(256), 0, stream>>>(
        src, dst, total, cols, colsPad);
  };
  // select register-blocked variant: 32x64 tile; 32x80 for N=80; K is static
  auto gemm = [&](const bf16_t* A, const bf16_t* W, const float* bias, const float* res,
                  float* C, int M, int N, int Kp, int act) {
    if ((N % 64 == 0) && (M % 32 == 0)) {
      int groups = (M / 32) * (N / 64);
      dim3 grid((unsigned)((groups + 7) / 8)), blk(256);
      switch (Kp) {
        case 32:   wmma_gemm_kernel<2, 4, 32>  <<<grid, blk, 0, stream>>>(A, W, bias, res, C, M, N, act); return;
        case 64:   wmma_gemm_kernel<2, 4, 64>  <<<grid, blk, 0, stream>>>(A, W, bias, res, C, M, N, act); return;
        case 256:  wmma_gemm_kernel<2, 4, 256> <<<grid, blk, 0, stream>>>(A, W, bias, res, C, M, N, act); return;
        case 768:  wmma_gemm_kernel<2, 4, 768> <<<grid, blk, 0, stream>>>(A, W, bias, res, C, M, N, act); return;
        case 1536: wmma_gemm_kernel<2, 4, 1536><<<grid, blk, 0, stream>>>(A, W, bias, res, C, M, N, act); return;
        default: break;
      }
    } else if ((N == 80) && (M % 32 == 0) && (Kp == 1536)) {
      int groups = (M / 32) * (N / 80);
      dim3 grid((unsigned)((groups + 7) / 8)), blk(256);
      wmma_gemm_kernel<2, 5, 1536><<<grid, blk, 0, stream>>>(A, W, bias, res, C, M, N, act);
      return;
    }
    int groups = (M / 16) * (N / 16);
    wmma_gemm_dyn<<<dim3((unsigned)((groups + 7) / 8)), dim3(256), 0, stream>>>(
        A, W, bias, res, C, M, N, Kp, act);
  };

  // ---- weight + input conversions (bf16, with zero padding where needed) ----
  cvt(x,      x_bf,      MROWS,          DIN_, DINP_);
  cvt(w1,     w1_bf,     256,            DIN_, DINP_);
  cvt(w2,     w2_bf,     256,            256,  256);
  cvt(w3,     w3_bf,     H_,             256,  256);
  cvt(inw,    inw_bf,    (size_t)NL_ * I2,   H_,   H_);
  cvt(xprojw, xprojw_bf, (size_t)NL_ * SSMW, ICH,  ICH);
  cvt(dtw,    dtw_bf,    (size_t)NL_ * ICH,  RR,   RPAD);
  cvt(outw,   outw_bf,   (size_t)NL_ * H_,   ICH,  ICH);
  cvt(aw1,    aw1_bf,    DA_,            H_,   H_);

  // ---- input MLP: 13->256->256->768 (WMMA) ----
  gemm(x_bf,     w1_bf, b1, nullptr, t256a, MROWS, 256, DINP_, ACT_RELU);
  cvt(t256a, t256a_bf, MROWS, 256, 256);
  gemm(t256a_bf, w2_bf, b2, nullptr, t256b, MROWS, 256, 256,   ACT_RELU);
  cvt(t256b, t256b_bf, MROWS, 256, 256);
  gemm(t256b_bf, w3_bf, b3, nullptr, h,     MROWS, H_,  256,   ACT_NONE);

  // ---- Mamba layers ----
  const int nElemIC = MROWS * ICH;
  for (int l = 0; l < NL_; ++l) {
    rmsnorm_kernel<<<dim3(MROWS), dim3(256), 0, stream>>>(h, norm_w + (size_t)l * H_,
                                                          hn_bf, nullptr, H_);
    gemm(hn_bf, inw_bf + (size_t)l * I2 * H_, nullptr, nullptr, proj, MROWS, I2, H_, ACT_NONE);
    conv_silu_kernel<<<dim3((nElemIC + 255) / 256), dim3(256), 0, stream>>>(
        proj, convw + (size_t)l * ICH * KC, convb + (size_t)l * ICH, u, u_bf);
    gemm(u_bf, xprojw_bf + (size_t)l * SSMW * ICH, nullptr, nullptr, ssm, MROWS, SSMW, ICH, ACT_NONE);
    ts_pad_kernel<<<dim3((MROWS * RPAD + 255) / 256), dim3(256), 0, stream>>>(ssm, ts_bf);
    gemm(ts_bf, dtw_bf + (size_t)l * ICH * RPAD, dtb + (size_t)l * ICH, nullptr,
         dt, MROWS, ICH, RPAD, ACT_SOFTPLUS);
    scan_kernel<<<dim3((B_ * ICH + 255) / 256), dim3(256), 0, stream>>>(
        dt, u, ssm, alog + (size_t)l * ICH * NST, Dp + (size_t)l * ICH, y);
    gate_mul_kernel<<<dim3((nElemIC + 255) / 256), dim3(256), 0, stream>>>(y, proj, y_bf);
    gemm(y_bf, outw_bf + (size_t)l * H_ * ICH, nullptr, h /*residual*/, h, MROWS, H_, ICH, ACT_NONE);
  }

  // ---- head: rmsnorm, attention pooling, output MLP ----
  rmsnorm_kernel<<<dim3(MROWS), dim3(256), 0, stream>>>(h, normf, hf_bf, hf, H_);
  gemm(hf_bf, aw1_bf, nullptr, nullptr, s1, MROWS, DA_, H_, ACT_TANH);
  small_gemm_kernel<<<dim3((MROWS * RP_ + 255) / 256), dim3(256), 0, stream>>>(
      s1, aw2, nullptr, logits, MROWS, RP_, DA_, ACT_NONE);
  softmax_l_kernel<<<dim3(B_ * RP_), dim3(256), 0, stream>>>(logits, attnw);
  pool_kernel<<<dim3((B_ * RP_ * H_ + 255) / 256), dim3(256), 0, stream>>>(attnw, hf, mpool);
  small_gemm_kernel<<<dim3((B_ * 256 + 255) / 256), dim3(256), 0, stream>>>(
      mpool, ow1, ob1, o1, B_, 256, RP_ * H_, ACT_RELU);
  small_gemm_kernel<<<dim3((B_ * 256 + 255) / 256), dim3(256), 0, stream>>>(
      o1, ow2, ob2, o2, B_, 256, 256, ACT_RELU);
  small_gemm_kernel<<<dim3(1), dim3(256), 0, stream>>>(
      o2, ow3, ob3, out, B_, 1, 256, ACT_NONE);
}
